// kws_bnn_65824668778645
// MI455X (gfx1250) — 8-bit matrix operand order repaired and checked by reading, compiled, not run on hardware
//
#include <hip/hip_runtime.h>
#include <hip/hip_bf16.h>


typedef int v8i __attribute__((ext_vector_type(8)));

#define WAVES_PER_BLOCK 8
#define TILE 16
#define BN_EPS 1e-5f

__global__ __launch_bounds__(WAVES_PER_BLOCK * 32)
void bnn_fused_kernel(const float* __restrict__ x,
                      const float* __restrict__ conv_w,
                      const float* __restrict__ conv_b,
                      const float* __restrict__ bn2_g, const float* __restrict__ bn2_b,
                      const float* __restrict__ bn2_m, const float* __restrict__ bn2_v,
                      const float* __restrict__ fc_w,
                      const float* __restrict__ fc_b,
                      const float* __restrict__ bn1_g, const float* __restrict__ bn1_b,
                      const float* __restrict__ bn1_m, const float* __restrict__ bn1_v,
                      float* __restrict__ out,
                      int nImg)
{
    __shared__ unsigned   rowbits[WAVES_PER_BLOCK][TILE][28];
    __shared__ signed char act[WAVES_PER_BLOCK][TILE][128];
    __shared__ int        zbuf[WAVES_PER_BLOCK][TILE][16];
    __shared__ unsigned    wmask[5][6];
    __shared__ float       cA[5], cB[5];
    __shared__ signed char fcsgn[16][128];

    const int tid  = threadIdx.x;
    const int wave = tid >> 5;
    const int lane = tid & 31;

    for (int idx = tid; idx < 16 * 128; idx += blockDim.x) {
        int n = idx >> 7, k = idx & 127;
        signed char s = 0;
        if (n < 10 && k < 125) s = (fc_w[n * 125 + k] >= 0.f) ? (signed char)1 : (signed char)-1;
        fcsgn[n][k] = s;
    }
    for (int idx = tid; idx < 30; idx += blockDim.x) {
        int ch = idx / 6, kh = idx % 6;
        unsigned m = 0;
        for (int kw = 0; kw < 6; ++kw)
            m |= (conv_w[ch * 36 + kh * 6 + kw] >= 0.f ? 1u : 0u) << kw;
        wmask[ch][kh] = m;
    }
    for (int ch = tid; ch < 5; ch += blockDim.x) {
        float sc = bn2_g[ch] * rsqrtf(bn2_v[ch] + BN_EPS);
        cA[ch] = sc;
        cB[ch] = (conv_b[ch] - bn2_m[ch]) * sc + bn2_b[ch];
    }
    __syncthreads();

    const int tileBase = (blockIdx.x * WAVES_PER_BLOCK + wave) * TILE;

    for (int i = 0; i < (TILE * 28) / 32; ++i) {
        int pair = i * 32 + lane;
        int img = pair / 28, r = pair % 28;
        if (tileBase + img < nImg) {
            const float4* rp = reinterpret_cast<const float4*>(
                x + (size_t)(tileBase + img) * 784 + (size_t)r * 28);
            unsigned m = 0;
#pragma unroll
            for (int j = 0; j < 7; ++j) {
                float4 v = rp[j];
                m |= (v.x >= 0.f ? 1u : 0u) << (4 * j);
                m |= (v.y >= 0.f ? 1u : 0u) << (4 * j + 1);
                m |= (v.z >= 0.f ? 1u : 0u) << (4 * j + 2);
                m |= (v.w >= 0.f ? 1u : 0u) << (4 * j + 3);
            }
            rowbits[wave][img][r] = m;
        }
    }
    __syncthreads();

    for (int i = 0; i < 63; ++i) {
        int f = i * 32 + lane;
        if (f < TILE * 125) {
            int img = f / 125;
            int k   = f % 125;
            int ch  = k / 25;
            int pos = k % 25;
            int oh = pos / 5, ow = pos % 5;
            int sh = oh * 6 - 2, sw = ow * 6 - 2;
            int kh0 = (oh == 0) ? 2 : 0;
            unsigned colmask = (ow == 0) ? 0x3Cu : 0x3Fu;
            int ncols = (ow == 0) ? 4 : 6;
            int agree = 0, total = 0;
            for (int kh = kh0; kh < 6; ++kh) {
                unsigned xrow = rowbits[wave][img][sh + kh];
                unsigned xm = (sw >= 0) ? ((xrow >> sw) & 0x3Fu)
                                        : ((xrow << 2) & 0x3Fu);
                unsigned wm = wmask[ch][kh];
                agree += __popc((~(xm ^ wm)) & colmask);
                total += ncols;
            }
            int dot = 2 * agree - total;
            float s = (float)dot * cA[ch] + cB[ch];
            act[wave][img][k] = (s >= 0.f) ? (signed char)1 : (signed char)-1;
        }
    }
    for (int i = 0; i < 2; ++i) {
        int e = i * 32 + lane;
        if (e < TILE * 3) act[wave][e / 3][125 + e % 3] = 0;
    }
    __syncthreads();

    const int row = lane & 15;
    const int hi  = (lane >> 4) & 1;
    v8i acc = {0, 0, 0, 0, 0, 0, 0, 0};
#pragma unroll
    for (int c = 0; c < 2; ++c) {
        v8i A, Bm;
#pragma unroll
        for (int j = 0; j < 8; ++j) {
            int kbase = c * 64 + (j >> 1) * 16 + hi * 8 + (j & 1) * 4;
            A[j] = *reinterpret_cast<const int*>(&act[wave][row][kbase]);
            int kbb = c * 64 + (j >> 1) * 16 + hi * 8 + (j & 1) * 4;
            unsigned b0 = (unsigned char)fcsgn[row][kbb + 0];
            unsigned b1 = (unsigned char)fcsgn[row][kbb + 1];
            unsigned b2 = (unsigned char)fcsgn[row][kbb + 2];
            unsigned b3 = (unsigned char)fcsgn[row][kbb + 3];
            Bm[j] = (int)(b0 | (b1 << 8) | (b2 << 16) | (b3 << 24));
        }
        acc = __builtin_amdgcn_wmma_i32_16x16x64_iu8(true, A, true, Bm, acc,
                                                      false,  false);
    }

#pragma unroll
    for (int r = 0; r < 8; ++r)
        zbuf[wave][r + hi * 8][row] = acc[r];
    __syncthreads();

    if (lane < 16) {
        int img = tileBase + lane;
        if (img < nImg) {
            float t[10];
            float mx = -1e30f;
#pragma unroll
            for (int n = 0; n < 10; ++n) {
                float z = (float)zbuf[wave][lane][n] + fc_b[n];
                z = (z - bn1_m[n]) * (bn1_g[n] * rsqrtf(bn1_v[n] + BN_EPS)) + bn1_b[n];
                t[n] = z;
                mx = fmaxf(mx, z);
            }
            float se = 0.f;
#pragma unroll
            for (int n = 0; n < 10; ++n) se += __expf(t[n] - mx);
            float lse = mx + __logf(se);
            float* op = out + (size_t)img * 10;
#pragma unroll
            for (int n = 0; n < 10; ++n) op[n] = t[n] - lse;
        }
    }
}

extern "C" void kernel_launch(void* const* d_in, const int* in_sizes, int n_in,
                              void* d_out, int out_size, void* d_ws, size_t ws_size,
                              hipStream_t stream) {
    const float* x      = (const float*)d_in[0];
    const float* conv_w = (const float*)d_in[1];
    const float* conv_b = (const float*)d_in[2];
    const float* bn2_g  = (const float*)d_in[3];
    const float* bn2_b  = (const float*)d_in[4];
    const float* bn2_m  = (const float*)d_in[5];
    const float* bn2_v  = (const float*)d_in[6];
    const float* fc_w   = (const float*)d_in[7];
    const float* fc_b   = (const float*)d_in[8];
    const float* bn1_g  = (const float*)d_in[9];
    const float* bn1_b  = (const float*)d_in[10];
    const float* bn1_m  = (const float*)d_in[11];
    const float* bn1_v  = (const float*)d_in[12];
    float* out = (float*)d_out;

    int nImg = in_sizes[0] / 784;
    int imgsPerBlock = WAVES_PER_BLOCK * TILE;
    int grid = (nImg + imgsPerBlock - 1) / imgsPerBlock;

    bnn_fused_kernel<<<grid, WAVES_PER_BLOCK * 32, 0, stream>>>(
        x, conv_w, conv_b, bn2_g, bn2_b, bn2_m, bn2_v,
        fc_w, fc_b, bn1_g, bn1_b, bn1_m, bn1_v, out, nImg);
}
